// GTRANRel_1666447311028
// MI455X (gfx1250) — matrix operand order repaired and checked by reading, not compiled, not run on hardware
//
#include <hip/hip_runtime.h>
#include <hip/hip_bf16.h>

#define FDIM   128
#define HEADS  4
#define DHEAD  32
#define SCALEF 0.1767766952966369f  /* 1/sqrt(32) */

typedef __attribute__((ext_vector_type(16))) __bf16 v16bf;
typedef __attribute__((ext_vector_type(8)))  __bf16 v8bf;
typedef __attribute__((ext_vector_type(8)))  float  v8f;

// ---------------------------------------------------------------- utilities
__global__ void fill_f32(float* __restrict__ p, float v, int n) {
    int i = blockIdx.x * blockDim.x + threadIdx.x;
    if (i < n) p[i] = v;
}

// monotonic uint key for float atomicMax
__device__ __forceinline__ unsigned fkey(float f) {
    unsigned u = __float_as_uint(f);
    return (u & 0x80000000u) ? ~u : (u | 0x80000000u);
}
__device__ __forceinline__ float fkey_dec(unsigned u) {
    return (u & 0x80000000u) ? __uint_as_float(u & 0x7fffffffu)
                             : __uint_as_float(~u);
}

// ------------------------------------------------- one-time bf16 repacking
// W (f32, [k][col] row-major) -> Wt (bf16, [col][k]) : makes each lane's
// 16 B-operand values contiguous (32B) in the WMMA inner loop.
__global__ void w_transpose_bf16(const float* __restrict__ W,
                                 __bf16* __restrict__ Wt) {
    int i = blockIdx.x * blockDim.x + threadIdx.x;
    if (i >= FDIM * FDIM) return;
    int col = i >> 7, k = i & 127;
    Wt[(size_t)col * FDIM + k] = (__bf16)W[(size_t)k * FDIM + col];
}

__global__ void h_to_bf16(const float* __restrict__ h,
                          __bf16* __restrict__ hb, int n) {
    int i = blockIdx.x * blockDim.x + threadIdx.x;
    if (i < n) hb[i] = (__bf16)h[i];
}

// ------------------------------------------------- relation-type projection
// re[l,r,:] = rel_emb[r,:] @ We[l] + be[l]   (collapses E x 128 x 128 GEMM)
__global__ void rel_proj_kernel(const float* __restrict__ rel_emb,
                                const float* __restrict__ We,
                                const float* __restrict__ be,
                                float* __restrict__ re, int Rn, int Ln) {
    int i = blockIdx.x * blockDim.x + threadIdx.x;
    int total = Ln * Rn * FDIM;
    if (i >= total) return;
    int j = i % FDIM;
    int r = (i / FDIM) % Rn;
    int l = i / (FDIM * Rn);
    const float* w  = We + (size_t)l * FDIM * FDIM;
    const float* em = rel_emb + (size_t)r * FDIM;
    float acc = be[l * FDIM + j];
    for (int t = 0; t < FDIM; ++t) acc += em[t] * w[t * FDIM + j];
    re[i] = acc;
}

// ------------------------------------------------------- fused QKV WMMA GEMM
// One 16-row tile per block; 8 waves cover the 128 output columns.
// All operands pre-converted to bf16: per K-step = 2 A loads (16B) +
// 3x1 B loads (32B) + 3 WMMAs, zero cvts in the loop.
__global__ __launch_bounds__(256) void qkv_gemm_wmma(
    const __bf16* __restrict__ hb,
    const __bf16* __restrict__ WqT, const __bf16* __restrict__ WkT,
    const __bf16* __restrict__ WvT,
    const float* __restrict__ bq, const float* __restrict__ bk,
    const float* __restrict__ bv,
    float* __restrict__ qo, float* __restrict__ ko, float* __restrict__ vo,
    int Nn) {
    const int wave = threadIdx.x >> 5;
    const int lane = threadIdx.x & 31;
    const int rl   = lane & 15;
    const int hi   = lane >> 4;              // half-wave select
    const int row0 = blockIdx.x * 16;
    int arow = row0 + rl;                    // A-tile row for this lane
    if (arow >= Nn) arow = Nn - 1;           // clamp: keep EXEC all-ones
    const int col = wave * 16 + rl;          // B/D column for this lane

    v8f aq = {}, ak = {}, av = {};
    const __bf16* hrow = hb  + (size_t)arow * FDIM;
    const __bf16* bqc  = WqT + (size_t)col * FDIM;
    const __bf16* bkc  = WkT + (size_t)col * FDIM;
    const __bf16* bvc  = WvT + (size_t)col * FDIM;

    #pragma unroll
    for (int kk = 0; kk < FDIM; kk += 32) {
        // A 16x32 bf16: elem i -> K = kk + 16*hi + i
        v8bf a0 = *(const v8bf*)(hrow + kk + hi * 16);
        v8bf a1 = *(const v8bf*)(hrow + kk + hi * 16 + 8);
        v16bf A = __builtin_shufflevector(a0, a1, 0, 1, 2, 3, 4, 5, 6, 7,
                                          8, 9, 10, 11, 12, 13, 14, 15);
        // B 32x16 bf16: elem i -> K = kk + 16*hi + i, column = col
        const int kb = kk + hi * 16;
        v16bf Bq = *(const v16bf*)(bqc + kb);
        v16bf Bk = *(const v16bf*)(bkc + kb);
        v16bf Bv = *(const v16bf*)(bvc + kb);

        aq = __builtin_amdgcn_wmma_f32_16x16x32_bf16(false, A, false, Bq,
                                                     (short)0, aq, false, false);
        ak = __builtin_amdgcn_wmma_f32_16x16x32_bf16(false, A, false, Bk,
                                                     (short)0, ak, false, false);
        av = __builtin_amdgcn_wmma_f32_16x16x32_bf16(false, A, false, Bv,
                                                     (short)0, av, false, false);
    }

    const float biasq = bq[col], biask = bk[col], biasv = bv[col];
    #pragma unroll
    for (int r = 0; r < 8; ++r) {            // D elem r -> row = r + 8*hi
        int orow = row0 + r + hi * 8;
        if (orow < Nn) {
            size_t off = (size_t)orow * FDIM + col;
            qo[off] = aq[r] + biasq;
            ko[off] = ak[r] + biask;
            vo[off] = av[r] + biasv;
        }
    }
}

// ------------------------------------------------------------- edge kernels
__global__ void edge_score_kernel(const float* __restrict__ q,
                                  const float* __restrict__ k,
                                  const float* __restrict__ re_l,
                                  const int* __restrict__ src,
                                  const int* __restrict__ dst,
                                  const int* __restrict__ et,
                                  float* __restrict__ score,
                                  unsigned* __restrict__ mkey, int En) {
    int i = blockIdx.x * blockDim.x + threadIdx.x;
    if (i >= En * HEADS) return;
    int e = i >> 2, hh = i & 3;
    int s = src[e], d = dst[e], t = et[e];
    const float* qp = q    + (size_t)d * FDIM + hh * DHEAD;
    const float* kp = k    + (size_t)s * FDIM + hh * DHEAD;
    const float* ep = re_l + (size_t)t * FDIM + hh * DHEAD;
    float acc = 0.f;
    #pragma unroll
    for (int x = 0; x < DHEAD; ++x) acc += qp[x] * (kp[x] + ep[x]);
    acc *= SCALEF;
    score[i] = acc;
    atomicMax(&mkey[(size_t)d * HEADS + hh], fkey(acc));
}

__global__ void edge_exp_kernel(const int* __restrict__ dst,
                                const unsigned* __restrict__ mkey,
                                float* __restrict__ score_ex,
                                float* __restrict__ ssum, int En) {
    int i = blockIdx.x * blockDim.x + threadIdx.x;
    if (i >= En * HEADS) return;
    int e = i >> 2, hh = i & 3;
    int d = dst[e];
    float m  = fkey_dec(mkey[(size_t)d * HEADS + hh]);
    float ex = expf(score_ex[i] - m);
    score_ex[i] = ex;
    atomicAdd(&ssum[(size_t)d * HEADS + hh], ex);
}

__global__ void edge_agg_kernel(const float* __restrict__ ex,
                                const float* __restrict__ ssum,
                                const float* __restrict__ v,
                                const float* __restrict__ re_l,
                                const int* __restrict__ src,
                                const int* __restrict__ dst,
                                const int* __restrict__ et,
                                float* __restrict__ agg, int En) {
    int i = blockIdx.x * blockDim.x + threadIdx.x;
    if (i >= En * HEADS) return;
    int e = i >> 2, hh = i & 3;
    int s = src[e], d = dst[e], t = et[e];
    float attn = ex[i] / (ssum[(size_t)d * HEADS + hh] + 1e-9f);
    const float* vp = v    + (size_t)s * FDIM + hh * DHEAD;
    const float* ep = re_l + (size_t)t * FDIM + hh * DHEAD;
    float* ap       = agg  + (size_t)d * FDIM + hh * DHEAD;
    #pragma unroll
    for (int x = 0; x < DHEAD; ++x) atomicAdd(&ap[x], attn * (vp[x] + ep[x]));
}

// residual + ELU, fused with bf16 mirror update for the next layer's GEMM
__global__ void residual_elu_kernel(const float* __restrict__ agg,
                                    const float* __restrict__ hin,
                                    float* __restrict__ hout,
                                    __bf16* __restrict__ hb, int n) {
    int i = blockIdx.x * blockDim.x + threadIdx.x;
    if (i >= n) return;
    float val = agg[i] + hin[i];
    float r = val > 0.f ? val : (expf(val) - 1.f);
    hout[i] = r;
    hb[i]   = (__bf16)r;
}

__global__ void output_kernel(const float* __restrict__ h,
                              const float* __restrict__ Wout,
                              const float* __restrict__ bout,
                              const float* __restrict__ cent,
                              const float* __restrict__ gamma,
                              const float* __restrict__ beta,
                              float* __restrict__ out, int Nn) {
    int n = blockIdx.x * blockDim.x + threadIdx.x;
    if (n >= Nn) return;
    const float* hp = h + (size_t)n * FDIM;
    float acc = bout[0];
    for (int i = 0; i < FDIM; ++i) acc += hp[i] * Wout[i];
    float sc = cent[n] * gamma[0] + beta[0];
    float l  = sc * acc;
    out[n] = l > 0.f ? l : 0.f;
}

// ------------------------------------------------------------------ launch
extern "C" void kernel_launch(void* const* d_in, const int* in_sizes, int n_in,
                              void* d_out, int out_size, void* d_ws,
                              size_t ws_size, hipStream_t stream) {
    const float* inputs     = (const float*)d_in[0];
    const int*   edge_types = (const int*)  d_in[1];
    const int*   src        = (const int*)  d_in[2];
    const int*   dst        = (const int*)  d_in[3];
    const float* centrality = (const float*)d_in[4];
    const float* rel_emb    = (const float*)d_in[5];
    const float* Wq         = (const float*)d_in[6];
    const float* bq         = (const float*)d_in[7];
    const float* Wk         = (const float*)d_in[8];
    const float* bk         = (const float*)d_in[9];
    const float* Wv         = (const float*)d_in[10];
    const float* bv         = (const float*)d_in[11];
    const float* We         = (const float*)d_in[12];
    const float* be         = (const float*)d_in[13];
    const float* Wout       = (const float*)d_in[14];
    const float* bout       = (const float*)d_in[15];
    const float* gamma      = (const float*)d_in[16];
    const float* beta       = (const float*)d_in[17];

    const int Nn = in_sizes[0] / FDIM;
    const int En = in_sizes[1];
    const int Rn = in_sizes[5] / FDIM;
    const int Ln = in_sizes[6] / (FDIM * FDIM);

    // workspace layout (f32-element units)
    float* ws = (float*)d_ws;
    size_t off = 0;
    float*    qb   = ws + off; off += (size_t)Nn * FDIM;
    float*    kb   = ws + off; off += (size_t)Nn * FDIM;
    float*    vb   = ws + off; off += (size_t)Nn * FDIM;
    float*    agg  = ws + off; off += (size_t)Nn * FDIM;
    float*    hbuf = ws + off; off += (size_t)Nn * FDIM;
    float*    exb  = ws + off; off += (size_t)En * HEADS;
    unsigned* mkey = (unsigned*)(ws + off); off += (size_t)Nn * HEADS;
    float*    ssum = ws + off; off += (size_t)Nn * HEADS;
    float*    reb  = ws + off; off += (size_t)Ln * Rn * FDIM;
    __bf16*   hbf  = (__bf16*)(ws + off); off += ((size_t)Nn * FDIM + 1) / 2;
    __bf16*   wpk  = (__bf16*)(ws + off); off += ((size_t)Ln * 3 * FDIM * FDIM + 1) / 2;

    const int WMAT = FDIM * FDIM;                 // 16384 elements per matrix
    const int WBLK = (WMAT + 255) / 256;

    // one-time: transpose+convert all projection weights to bf16 [col][k]
    for (int l = 0; l < Ln; ++l) {
        __bf16* WqT = wpk + (size_t)(l * 3 + 0) * WMAT;
        __bf16* WkT = wpk + (size_t)(l * 3 + 1) * WMAT;
        __bf16* WvT = wpk + (size_t)(l * 3 + 2) * WMAT;
        w_transpose_bf16<<<WBLK, 256, 0, stream>>>(Wq + (size_t)l * WMAT, WqT);
        w_transpose_bf16<<<WBLK, 256, 0, stream>>>(Wk + (size_t)l * WMAT, WkT);
        w_transpose_bf16<<<WBLK, 256, 0, stream>>>(Wv + (size_t)l * WMAT, WvT);
    }

    // relation projections for all layers (tiny GEMM + bias)
    {
        int tot = Ln * Rn * FDIM;
        rel_proj_kernel<<<(tot + 255) / 256, 256, 0, stream>>>(
            rel_emb, We, be, reb, Rn, Ln);
    }

    const int EH = En * HEADS;
    const int NF = Nn * FDIM;
    const int NH = Nn * HEADS;

    // bf16 mirror of layer-1 input features
    h_to_bf16<<<(NF + 255) / 256, 256, 0, stream>>>(inputs, hbf, NF);

    for (int l = 0; l < Ln; ++l) {
        const float* hin  = (l == 0) ? inputs : hbuf;
        const float* re_l = reb + (size_t)l * Rn * FDIM;
        const __bf16* WqT = wpk + (size_t)(l * 3 + 0) * WMAT;
        const __bf16* WkT = wpk + (size_t)(l * 3 + 1) * WMAT;
        const __bf16* WvT = wpk + (size_t)(l * 3 + 2) * WMAT;

        fill_f32<<<(NF + 255) / 256, 256, 0, stream>>>(agg, 0.f, NF);
        fill_f32<<<(NH + 255) / 256, 256, 0, stream>>>((float*)mkey, 0.f, NH);
        fill_f32<<<(NH + 255) / 256, 256, 0, stream>>>(ssum, 0.f, NH);

        qkv_gemm_wmma<<<(Nn + 15) / 16, 256, 0, stream>>>(
            hbf, WqT, WkT, WvT,
            bq + l * FDIM, bk + l * FDIM, bv + l * FDIM,
            qb, kb, vb, Nn);

        edge_score_kernel<<<(EH + 255) / 256, 256, 0, stream>>>(
            qb, kb, re_l, src, dst, edge_types, exb, mkey, En);
        edge_exp_kernel<<<(EH + 255) / 256, 256, 0, stream>>>(
            dst, mkey, exb, ssum, En);
        edge_agg_kernel<<<(EH + 255) / 256, 256, 0, stream>>>(
            exb, ssum, vb, re_l, src, dst, edge_types, agg, En);

        // writes f32 h for residual/output AND bf16 mirror for next GEMM
        residual_elu_kernel<<<(NF + 255) / 256, 256, 0, stream>>>(
            agg, hin, hbuf, hbf, NF);
    }

    output_kernel<<<(Nn + 255) / 256, 256, 0, stream>>>(
        hbuf, Wout, bout, centrality, gamma, beta, (float*)d_out, Nn);
}
